// BitSPPF_72353019069108
// MI455X (gfx1250) — matrix operand order repaired and checked by reading, not compiled, not run on hardware
//
#include <hip/hip_runtime.h>

// ---------------------------------------------------------------------------
// BitSPPF for MI455X (gfx1250): ternary-quantized 1x1 convs as bf16 WMMA GEMMs
// + bf16 SPPF maxpool chain. Wave32, v_wmma_f32_16x16x32_bf16, async LDS DMA.
// ---------------------------------------------------------------------------

typedef __attribute__((ext_vector_type(16))) __bf16 v16bf;
typedef __attribute__((ext_vector_type(8)))  float  v8f;

#define HW   1600   // 40*40
#define B_   32
#define C1   1024
#define HID  512
#define C2   1024
#define KCAT 2048
#define NBIN 4096

// ---- bf16 helpers (manual, RNE) -------------------------------------------
__device__ __forceinline__ unsigned short f2bf(float f) {
    unsigned int u = __float_as_uint(f);
    unsigned int r = u + 0x7FFFu + ((u >> 16) & 1u);
    return (unsigned short)(r >> 16);
}
__device__ __forceinline__ float bf2f(unsigned short s) {
    return __uint_as_float(((unsigned int)s) << 16);
}

// ---- CDNA5 async global->LDS copy (GV mode, 16B), tracked by ASYNCcnt ------
__device__ __forceinline__ void async_g2l_b128(unsigned int lds_byte_addr,
                                               const void* gaddr) {
    asm volatile("global_load_async_to_lds_b128 %0, %1, off"
                 :
                 : "v"(lds_byte_addr), "v"((unsigned long long)gaddr)
                 : "memory");
}
__device__ __forceinline__ void wait_asynccnt0() {
    asm volatile("s_wait_asynccnt 0x0" ::: "memory");
}

// ---- scratch init ----------------------------------------------------------
__global__ void init_zero_k(unsigned int* p, int n) {
    for (int i = blockIdx.x * blockDim.x + threadIdx.x; i < n; i += gridDim.x * blockDim.x)
        p[i] = 0u;
}

// ---- absmax reduction (nonneg float bits order like uint) ------------------
__global__ void absmax_k(const float* __restrict__ w, int n, unsigned int* slot) {
    float mx = 0.f;
    for (int i = blockIdx.x * blockDim.x + threadIdx.x; i < n; i += gridDim.x * blockDim.x)
        mx = fmaxf(mx, fabsf(w[i]));
    atomicMax(slot, __float_as_uint(mx));
}

// ---- histogram of |w| over [0, amax] ---------------------------------------
__global__ void hist_k(const float* __restrict__ w, int n,
                       const unsigned int* __restrict__ slot, int* __restrict__ hist) {
    float amax = __uint_as_float(*slot);
    float inv = (float)NBIN / fmaxf(amax, 1e-20f);
    for (int i = blockIdx.x * blockDim.x + threadIdx.x; i < n; i += gridDim.x * blockDim.x) {
        int bin = (int)(fabsf(w[i]) * inv);
        bin = bin < NBIN - 1 ? bin : NBIN - 1;
        atomicAdd(&hist[bin], 1);
    }
}

// ---- approximate median select (bin center) --------------------------------
__global__ void select_k(const int* __restrict__ h1, const int* __restrict__ h2,
                         const unsigned int* __restrict__ slots, float* __restrict__ scales,
                         int n1, int n2) {
    if (threadIdx.x != 0 || blockIdx.x != 0) return;
    const int* hs[2] = { h1, h2 };
    int ns[2] = { n1, n2 };
    for (int t = 0; t < 2; ++t) {
        float amax = __uint_as_float(slots[t]);
        int target = ns[t] >> 1;
        long long cum = 0; int bin = NBIN - 1;
        for (int j = 0; j < NBIN; ++j) { cum += hs[t][j]; if (cum >= target) { bin = j; break; } }
        float s = amax * ((float)bin + 0.5f) / (float)NBIN;
        scales[t] = fmaxf(s, 1e-8f);
    }
}

// ---- ternary quantization to bf16 {-1,0,1} ---------------------------------
__global__ void quant_k(const float* __restrict__ w, int n, const float* __restrict__ sptr,
                        unsigned short* __restrict__ t) {
    float s = *sptr;
    for (int i = blockIdx.x * blockDim.x + threadIdx.x; i < n; i += gridDim.x * blockDim.x) {
        float q = rintf(w[i] / s);
        t[i] = (q > 0.5f) ? 0x3F80u : ((q < -0.5f) ? 0xBF80u : 0u);
    }
}

// ---- fold scale*BN into per-channel affine ---------------------------------
__global__ void affine_k(const float* __restrict__ g, const float* __restrict__ be,
                         const float* __restrict__ m, const float* __restrict__ v,
                         const float* __restrict__ sptr, int n,
                         float* __restrict__ alpha, float* __restrict__ bias) {
    int o = blockIdx.x * blockDim.x + threadIdx.x;
    if (o >= n) return;
    float inv = g[o] * rsqrtf(v[o] + 1e-5f);
    alpha[o] = (*sptr) * inv;
    bias[o]  = be[o] - m[o] * inv;
}

// ---------------------------------------------------------------------------
// GEMM + BN + SiLU.  Block: M=512 out-channels x N=64 pixels, K stepped by 32.
// 8 waves; wave w owns rows [w*64, w*64+64), all 64 pixels.
// A = ternary weights (bf16) staged via async global->LDS DMA;
// B = activations (f32->bf16 or bf16) staged through VALU; acc f32.
// ---------------------------------------------------------------------------
template<int KTOT, bool B_IS_F32, bool OUT_BF16>
__global__ __launch_bounds__(256)
void gemm_bn_silu_k(const unsigned short* __restrict__ Aq,
                    const void* __restrict__ Bsrc,
                    const float* __restrict__ alpha,
                    const float* __restrict__ bias,
                    void* __restrict__ Out,
                    int outChStride)
{
    __shared__ unsigned short lA[512 * 32];   // [m][k]
    __shared__ unsigned short lB[64 * 32];    // [n][k]

    const int pBase = blockIdx.x * 64;
    const int mBase = blockIdx.y * 512;
    const int b     = blockIdx.z;

    const int tid  = threadIdx.x;
    const int wave = tid >> 5;
    const int lane = tid & 31;
    const int half = lane >> 4;
    const int ln16 = lane & 15;

    const float*          Bf = (const float*)Bsrc          + (size_t)b * KTOT * HW;
    const unsigned short* Bb = (const unsigned short*)Bsrc + (size_t)b * KTOT * HW;

    // LDS byte offset of lA (low 32 bits of generic pointer to shared memory)
    const unsigned int lA_base = (unsigned int)(unsigned long long)(&lA[0]);

    v8f acc[4][4] = {};

    for (int kt = 0; kt < KTOT / 32; ++kt) {
        __syncthreads();   // previous tile's DS reads drained

        // --- stage A tile: 512x32 bf16 via async DMA, 16B chunks ---
        #pragma unroll
        for (int i = 0; i < 8; ++i) {
            int q  = tid + i * 256;       // 0..2047 16B chunks
            int m  = q >> 2;
            int kk = (q & 3) << 3;
            const void* g = (const void*)(Aq + (size_t)(mBase + m) * KTOT + kt * 32 + kk);
            async_g2l_b128(lA_base + (unsigned int)(m * 32 + kk) * 2u, g);
        }
        // --- stage B tile: 32x64 -> LDS [n][k], convert if f32 ---
        #pragma unroll
        for (int i = 0; i < 8; ++i) {
            int e = tid + i * 256;        // 0..2047 elements
            int k = e >> 6;
            int n = e & 63;
            unsigned short bv;
            if (B_IS_F32) bv = f2bf(Bf[(size_t)(kt * 32 + k) * HW + pBase + n]);
            else          bv = Bb[(size_t)(kt * 32 + k) * HW + pBase + n];
            lB[n * 32 + k] = bv;
        }
        wait_asynccnt0();  // A-tile DMA landed in LDS
        __syncthreads();   // + B-tile stores visible to all waves

        // --- build fragments per ISA VGPR layouts ---
        v16bf afr[4], bfr[4];
        #pragma unroll
        for (int fm = 0; fm < 4; ++fm) {
            union { v16bf v; unsigned int u[8]; } U;
            int m = wave * 64 + fm * 16 + ln16;
            #pragma unroll
            for (int r = 0; r < 8; ++r) {
                int k0 = 2 * r + ((r >= 4) ? 8 : 0) + half * 8;   // A 16x32 bf16 map
                U.u[r] = *(const unsigned int*)&lA[m * 32 + k0];
            }
            afr[fm] = U.v;
        }
        #pragma unroll
        for (int fn = 0; fn < 4; ++fn) {
            union { v16bf v; unsigned int u[8]; } U;
            int n = fn * 16 + ln16;
            #pragma unroll
            for (int r = 0; r < 8; ++r) {
                int k0 = 2 * r + ((r >= 4) ? 8 : 0) + half * 8;   // B 32x16 bf16 map
                U.u[r] = *(const unsigned int*)&lB[n * 32 + k0];
            }
            bfr[fn] = U.v;
        }
        #pragma unroll
        for (int fm = 0; fm < 4; ++fm)
            #pragma unroll
            for (int fn = 0; fn < 4; ++fn)
                acc[fm][fn] = __builtin_amdgcn_wmma_f32_16x16x32_bf16(
                    false, afr[fm], false, bfr[fn], (short)0, acc[fm][fn], false, false);
    }

    // --- epilogue: affine + SiLU, scatter per C/D layout ---
    unsigned short* outB = (unsigned short*)Out;
    float*          outF = (float*)Out;
    #pragma unroll
    for (int fm = 0; fm < 4; ++fm) {
        #pragma unroll
        for (int fn = 0; fn < 4; ++fn) {
            #pragma unroll
            for (int r = 0; r < 8; ++r) {
                int mg = mBase + wave * 64 + fm * 16 + r + half * 8;
                int n  = pBase + fn * 16 + ln16;
                float y = acc[fm][fn][r] * alpha[mg] + bias[mg];
                y = y / (1.f + __expf(-y));
                size_t o = ((size_t)b * outChStride + mg) * HW + n;
                if (OUT_BF16) outB[o] = f2bf(y);
                else          outF[o] = y;
            }
        }
    }
}

// ---- 5x5 stride-1 pad-2 maxpool on bf16, within the 2048-channel cat buffer
__global__ void maxpool5_k(const unsigned short* __restrict__ cat_src,
                           unsigned short* __restrict__ cat_dst,
                           int srcC0, int dstC0) {
    int idx = blockIdx.x * blockDim.x + threadIdx.x;
    const int total = B_ * HID * HW;
    if (idx >= total) return;
    int p = idx % HW;
    int t = idx / HW;
    int c = t % HID;
    int b = t / HID;
    int hh = p / 40, ww = p % 40;
    const unsigned short* s = cat_src + ((size_t)b * KCAT + srcC0 + c) * HW;
    float mx = -INFINITY;
    #pragma unroll
    for (int dy = -2; dy <= 2; ++dy) {
        int y = hh + dy;
        if (y < 0 || y >= 40) continue;
        int base = y * 40;
        #pragma unroll
        for (int dx = -2; dx <= 2; ++dx) {
            int x = ww + dx;
            if (x < 0 || x >= 40) continue;
            mx = fmaxf(mx, bf2f(s[base + x]));
        }
    }
    cat_dst[((size_t)b * KCAT + dstC0 + c) * HW + p] = f2bf(mx);
}

// ---------------------------------------------------------------------------
// Workspace layout (byte offsets, 256-aligned)
// ---------------------------------------------------------------------------
#define WS_MAX    0u           // 2 * uint
#define WS_SCALE  256u         // 2 * float
#define WS_HIST   512u         // 2 * 4096 * int = 32768
#define WS_A1     33280u       // alpha1: 512 f
#define WS_B1     35328u       // bias1 : 512 f
#define WS_A2     37376u       // alpha2: 1024 f
#define WS_B2     41472u       // bias2 : 1024 f
#define WS_T1     45568u       // 512*1024 bf16  = 1 MB
#define WS_T2     1094144u     // 1024*2048 bf16 = 4 MB
#define WS_CAT    5288448u     // 32*2048*1600 bf16 = 209,715,200 B

extern "C" void kernel_launch(void* const* d_in, const int* in_sizes, int n_in,
                              void* d_out, int out_size, void* d_ws, size_t ws_size,
                              hipStream_t stream) {
    const float* x  = (const float*)d_in[0];
    const float* w1 = (const float*)d_in[1];
    const float* g1 = (const float*)d_in[2];
    const float* b1 = (const float*)d_in[3];
    const float* m1 = (const float*)d_in[4];
    const float* v1 = (const float*)d_in[5];
    const float* w2 = (const float*)d_in[6];
    const float* g2 = (const float*)d_in[7];
    const float* b2 = (const float*)d_in[8];
    const float* m2 = (const float*)d_in[9];
    const float* v2 = (const float*)d_in[10];

    char* ws = (char*)d_ws;
    unsigned int* maxslots = (unsigned int*)(ws + WS_MAX);
    float*  scales = (float*)(ws + WS_SCALE);
    int*    hist1  = (int*)(ws + WS_HIST);
    int*    hist2  = hist1 + NBIN;
    float*  alpha1 = (float*)(ws + WS_A1);
    float*  bias1  = (float*)(ws + WS_B1);
    float*  alpha2 = (float*)(ws + WS_A2);
    float*  bias2  = (float*)(ws + WS_B2);
    unsigned short* t1  = (unsigned short*)(ws + WS_T1);
    unsigned short* t2  = (unsigned short*)(ws + WS_T2);
    unsigned short* cat = (unsigned short*)(ws + WS_CAT);

    const int n1 = in_sizes[1];   // 512*1024
    const int n2 = in_sizes[6];   // 1024*2048

    // 0) zero accumulation scratch (maxslots + both histograms) every call
    init_zero_k<<<33, 256, 0, stream>>>((unsigned int*)(ws), 2 + 2 * NBIN + 62);

    // 1) median scales via absmax + histogram + select
    absmax_k<<<512, 256, 0, stream>>>(w1, n1, &maxslots[0]);
    absmax_k<<<1024, 256, 0, stream>>>(w2, n2, &maxslots[1]);
    hist_k<<<512, 256, 0, stream>>>(w1, n1, &maxslots[0], hist1);
    hist_k<<<1024, 256, 0, stream>>>(w2, n2, &maxslots[1], hist2);
    select_k<<<1, 32, 0, stream>>>(hist1, hist2, maxslots, scales, n1, n2);

    // 2) ternary quantize weights to bf16 {-1,0,1}
    quant_k<<<512, 256, 0, stream>>>(w1, n1, &scales[0], t1);
    quant_k<<<1024, 256, 0, stream>>>(w2, n2, &scales[1], t2);

    // 3) fold scale * BN into per-channel affine
    affine_k<<<(HID + 255) / 256, 256, 0, stream>>>(g1, b1, m1, v1, &scales[0], HID, alpha1, bias1);
    affine_k<<<(C2 + 255) / 256, 256, 0, stream>>>(g2, b2, m2, v2, &scales[1], C2, alpha2, bias2);

    // 4) GEMM1: h = SiLU(BN(t1 @ x)) -> cat channels [0,512), bf16
    {
        dim3 grid(HW / 64, HID / 512, B_);   // 25 x 1 x 32
        gemm_bn_silu_k<C1, true, true><<<grid, 256, 0, stream>>>(
            t1, (const void*)x, alpha1, bias1, (void*)cat, KCAT);
    }

    // 5) SPPF pool chain (bf16): ch [512,1024) <- [0,512) <- etc.
    {
        int total = B_ * HID * HW;
        int blocks = (total + 255) / 256;
        maxpool5_k<<<blocks, 256, 0, stream>>>(cat, cat, 0,    512);
        maxpool5_k<<<blocks, 256, 0, stream>>>(cat, cat, 512,  1024);
        maxpool5_k<<<blocks, 256, 0, stream>>>(cat, cat, 1024, 1536);
    }

    // 6) GEMM2: out = SiLU(BN(t2 @ cat)) -> d_out f32
    {
        dim3 grid(HW / 64, C2 / 512, B_);    // 25 x 2 x 32
        gemm_bn_silu_k<KCAT, false, false><<<grid, 256, 0, stream>>>(
            t2, (const void*)cat, alpha2, bias2, d_out, C2);
    }
}
